// Agent_81801947119731
// MI455X (gfx1250) — matrix operand order repaired and checked by reading, not compiled, not run on hardware
//
#include <hip/hip_runtime.h>
#include <hip/hip_bf16.h>
#include <stdint.h>

typedef __attribute__((ext_vector_type(16))) _Float16 v16h;
typedef __attribute__((ext_vector_type(8)))  _Float16 v8h;
typedef __attribute__((ext_vector_type(8)))  float    v8f;
typedef __attribute__((ext_vector_type(4)))  int      v4i;

// ---- CDNA5 async global->LDS (ASYNCcnt) support, compile-guarded ----------
// Builtin signature (from clang-22 diagnostic): param0 = AS1 (global) v4i*,
// param1 = AS3 (LDS) v4i*, then two integer immediates (offset, cpol).
#if defined(__HIP_DEVICE_COMPILE__) && \
    __has_builtin(__builtin_amdgcn_global_load_async_to_lds_b128)
#define HAVE_ASYNC_LDS 1
#define ASYNC_CP_B128(gsrc, ldst)                                             \
  __builtin_amdgcn_global_load_async_to_lds_b128(                             \
      (__attribute__((address_space(1))) v4i*)(uintptr_t)(gsrc),              \
      (__attribute__((address_space(3))) v4i*)(uintptr_t)(ldst), 0, 0)
#if __has_builtin(__builtin_amdgcn_s_wait_asynccnt)
#define ASYNC_WAIT0() __builtin_amdgcn_s_wait_asynccnt(0)
#else
#define ASYNC_WAIT0() asm volatile("s_wait_asynccnt 0x0" ::: "memory")
#endif
#else
#define HAVE_ASYNC_LDS 0
#endif

// ---------------------------------------------------------------------------
// WMMA fragment helpers (wave32, V_WMMA_F32_16X16X32_F16 layouts per ISA 7.12.2)
// A 16x32 f16: lane L (0-15): M=L, halves[0..7]=K 0..7, halves[8..15]=K 16..23
//              lane L (16-31): M=L-16, K 8..15 and 24..31
// B 32x16 f16: lane L: N=L&15, same K order as A (K 0..7,16..23 / 8..15,24..31)
// C/D 16x16 f32: VGPR r, lanes0-15 -> M=r, lanes16-31 -> M=8+r, N=lane&15
// ---------------------------------------------------------------------------
__device__ __forceinline__ v16h frag_from_chunks(v8h lo, v8h hi) {
  v16h r;
#pragma unroll
  for (int i = 0; i < 8; ++i) { r[i] = lo[i]; r[i + 8] = hi[i]; }
  return r;
}

__device__ __forceinline__ v16h load_fragA(const _Float16* __restrict__ A, int lda,
                                           int m0, int kk, int lane) {
  const _Float16* p = A + (long)(m0 + (lane & 15)) * lda + kk + ((lane >> 4) << 3);
  return frag_from_chunks(*(const v8h*)p, *(const v8h*)(p + 16));
}

__device__ __forceinline__ v16h load_fragB(const _Float16* __restrict__ B, int ldb,
                                           int n0, int kk, int lane) {
  const _Float16* p = B + (long)(n0 + (lane & 15)) * ldb + kk + ((lane >> 4) << 3);
  return frag_from_chunks(*(const v8h*)p, *(const v8h*)(p + 16));
}

// ---------------------------------------------------------------------------
// Generic GEMM:  C[M,N] = act(A[M,K](f16) * B[N,K](f16)^T + bias)
// One wave computes a 16 x (16*NT) register tile: the A fragment is loaded
// once per K-step and reused across NT B fragments (NT WMMAs per A load).
// 8 waves / 256-thread block.
// ---------------------------------------------------------------------------
template <int NT>
__global__ void k_gemm(const _Float16* __restrict__ A, int lda,
                       const _Float16* __restrict__ B, int ldb,
                       const float* __restrict__ bias,
                       float* __restrict__ outF, _Float16* __restrict__ outH,
                       int M, int N, int K, int act) {
  const int lane = threadIdx.x & 31;
  const int wave = threadIdx.x >> 5;
  const int tn_tiles = N / (16 * NT);
  const long tile = (long)blockIdx.x * (blockDim.x >> 5) + wave;
  const long total = (long)(M >> 4) * tn_tiles;
  if (tile >= total) return;
  const int tm = (int)(tile / tn_tiles) << 4;
  const int tn = (int)(tile % tn_tiles) * (16 * NT);

  v8f c[NT];
#pragma unroll
  for (int j = 0; j < NT; ++j) c[j] = (v8f){0.f, 0.f, 0.f, 0.f, 0.f, 0.f, 0.f, 0.f};

  for (int kk = 0; kk < K; kk += 32) {
    __builtin_prefetch(A + (long)(tm + (lane & 15)) * lda + kk + 128, 0, 0);
    const v16h a = load_fragA(A, lda, tm, kk, lane);
#pragma unroll
    for (int j = 0; j < NT; ++j) {
      const v16h b = load_fragB(B, ldb, tn + j * 16, kk, lane);
      c[j] = __builtin_amdgcn_wmma_f32_16x16x32_f16(false, a, false, b, (short)0,
                                                    c[j], false, false);
    }
  }
  const int col = lane & 15;
  const int rb = (lane >> 4) << 3;
#pragma unroll
  for (int j = 0; j < NT; ++j) {
    const int nc = tn + j * 16 + col;
    const float bv = bias ? bias[nc] : 0.f;
#pragma unroll
    for (int r = 0; r < 8; ++r) {
      const long row = tm + rb + r;
      float v = c[j][r] + bv;
      if (act) v = fmaxf(v, 0.f);
      if (outF) outF[row * N + nc] = v;
      if (outH) outH[row * N + nc] = (_Float16)v;
    }
  }
}

// ---------------------------------------------------------------------------
// Implicit-GEMM conv (VALID, NCHW weights pre-permuted; activations NHWC f16).
// RAW=true: conv1 path, reads fp32 pixels, scales by 1/255, K = c*64 + kh*8 + kw
// RAW=false: K' = (kh*KW + kw)*IC + ic  -> 8-contig K == 8-contig fp16 in NHWC
// A fragment (expensive im2col addressing) reused across NT B fragments.
// ---------------------------------------------------------------------------
template <bool RAW, int NT, int IC, int KW, int IH, int IW, int OH, int OW, int STR>
__global__ void k_conv(const void* __restrict__ inp, const _Float16* __restrict__ W,
                       const float* __restrict__ bias, _Float16* __restrict__ out,
                       int M, int N, int K) {
  const int lane = threadIdx.x & 31;
  const int wave = threadIdx.x >> 5;
  const int tn_tiles = N / (16 * NT);
  const long tile = (long)blockIdx.x * (blockDim.x >> 5) + wave;
  const long total = (long)(M >> 4) * tn_tiles;
  if (tile >= total) return;
  const int tm = (int)(tile / tn_tiles) << 4;
  const int tn = (int)(tile % tn_tiles) * (16 * NT);

  const int mA = tm + (lane & 15);
  const int n_img = mA / (OH * OW);
  const int sA = mA - n_img * (OH * OW);
  const int oh = sA / OW, ow = sA - oh * OW;

  v8f c[NT];
#pragma unroll
  for (int j = 0; j < NT; ++j) c[j] = (v8f){0.f, 0.f, 0.f, 0.f, 0.f, 0.f, 0.f, 0.f};

  for (int kk = 0; kk < K; kk += 32) {
    const int kb = kk + ((lane >> 4) << 3);
    v8h ch[2];
#pragma unroll
    for (int h = 0; h < 2; ++h) {
      const int kc = kb + h * 16;
      if constexpr (RAW) {  // conv1: 8x8 kernel, chunk = one kw row
        const int cc = kc >> 6;
        const int kh = (kc & 63) >> 3;
        const float* p = (const float*)inp + (long)n_img * (IC * IH * IW) +
                         (long)cc * (IH * IW) + (long)(oh * STR + kh) * IW + ow * STR;
        v8h t;
#pragma unroll
        for (int i = 0; i < 8; ++i) t[i] = (_Float16)(p[i] * (1.0f / 255.0f));
        ch[h] = t;
      } else {  // NHWC fp16 input, chunk = 8 contiguous channels
        const int p_ = kc / IC;
        const int ic0 = kc - p_ * IC;
        const int kh = p_ / KW, kw = p_ - kh * KW;
        const _Float16* p = (const _Float16*)inp +
                            ((long)n_img * (IH * IW) + (long)(oh * STR + kh) * IW +
                             (ow * STR + kw)) * IC + ic0;
        ch[h] = *(const v8h*)p;
      }
    }
    const v16h a = frag_from_chunks(ch[0], ch[1]);
#pragma unroll
    for (int j = 0; j < NT; ++j) {
      const v16h b = load_fragB(W, K, tn + j * 16, kk, lane);
      c[j] = __builtin_amdgcn_wmma_f32_16x16x32_f16(false, a, false, b, (short)0,
                                                    c[j], false, false);
    }
  }
  const int col = lane & 15;
  const int rb = (lane >> 4) << 3;
#pragma unroll
  for (int j = 0; j < NT; ++j) {
    const int nc = tn + j * 16 + col;
    const float bv = bias[nc];
#pragma unroll
    for (int r = 0; r < 8; ++r) {
      const int m2 = tm + rb + r;
      const int n2 = m2 / (OH * OW);
      const int s2 = m2 - n2 * (OH * OW);
      const float v = fmaxf(c[j][r] + bv, 0.f);
      out[((long)n2 * (OH * OW) + s2) * N + nc] = (_Float16)v;
    }
  }
}

// ------------------------- weight prep (fp32 -> fp16, permutes) -------------
__global__ void k_cvt(const float* __restrict__ s, _Float16* __restrict__ d, int n) {
  int i = blockIdx.x * 256 + threadIdx.x;
  if (i < n) d[i] = (_Float16)s[i];
}
__global__ void k_perm_c2(const float* __restrict__ s, _Float16* __restrict__ d) {
  int i = blockIdx.x * 256 + threadIdx.x;  // dst[oc][(kh*4+kw)*32+ic]
  if (i < 64 * 512) {
    int oc = i >> 9, k = i & 511, p = k >> 5, ic = k & 31;
    d[i] = (_Float16)s[oc * 512 + ic * 16 + p];
  }
}
__global__ void k_perm_c3(const float* __restrict__ s, _Float16* __restrict__ d) {
  int i = blockIdx.x * 256 + threadIdx.x;  // dst[oc][(kh*3+kw)*64+ic]
  if (i < 64 * 576) {
    int oc = i / 576, k = i - oc * 576, p = k >> 6, ic = k & 63;
    d[i] = (_Float16)s[oc * 576 + ic * 9 + p];
  }
}
__global__ void k_perm_fc(const float* __restrict__ s, _Float16* __restrict__ d) {
  int i = blockIdx.x * 256 + threadIdx.x;  // dst[D][(h*7+w)*64+c]
  if (i < 512 * 3136) {
    int dd = i / 3136, k = i - dd * 3136, hw = k >> 6, cc = k & 63;
    d[i] = (_Float16)s[(long)dd * 3136 + cc * 49 + hw];
  }
}

// ------------------- depthwise causal conv1d (K=4) + SiLU -------------------
__global__ void k_conv1d(const float* __restrict__ xz, const float* __restrict__ w,
                         const float* __restrict__ b, float* __restrict__ xcf,
                         _Float16* __restrict__ xch) {
  int idx = blockIdx.x * 256 + threadIdx.x;
  if (idx >= 2048 * 1024) return;
  int m = idx >> 10, d = idx & 1023;
  int t = m >> 4, bb = m & 15;
  float acc = b[d];
#pragma unroll
  for (int k = 0; k < 4; ++k) {
    int tt = t - 3 + k;
    if (tt >= 0) acc += w[d * 4 + k] * xz[((long)(tt * 16 + bb)) * 2048 + d];
  }
  float s = acc / (1.f + __expf(-acc));
  xcf[idx] = s;
  xch[idx] = (_Float16)s;
}

// --------------------- selective-scan (sequential in T) ---------------------
// Per timestep the 32-float [B16|C16] vector is staged into LDS. On CDNA5 we
// DMA it directly global->LDS with GLOBAL_LOAD_ASYNC_TO_LDS_B128 (ASYNCcnt),
// bypassing VGPRs; fallback is a plain register round-trip.
__global__ void k_ssm(const float* __restrict__ dbl, const float* __restrict__ dtraw,
                      const float* __restrict__ dtb, const float* __restrict__ alog,
                      const float* __restrict__ dskip, const float* __restrict__ xcf,
                      const float* __restrict__ xz, _Float16* __restrict__ yh) {
  const int bb = blockIdx.x;
  const int d = blockIdx.y * 256 + threadIdx.x;
  __shared__ float sBC[32];  // [0..15]=B, [16..31]=C
  float a[16], h[16];
#pragma unroll
  for (int n = 0; n < 16; ++n) {
    a[n] = -__expf(alog[(long)d * 16 + n]);
    h[n] = 0.f;
  }
  const float dtbv = dtb[d];
  const float dsk = dskip[d];
  for (int t = 0; t < 128; ++t) {
    const int m = t * 16 + bb;
#if HAVE_ASYNC_LDS
    if (threadIdx.x < 8)
      ASYNC_CP_B128(dbl + (long)m * 64 + 32 + threadIdx.x * 4,
                    &sBC[threadIdx.x * 4]);
    ASYNC_WAIT0();
#else
    if (threadIdx.x < 32) sBC[threadIdx.x] = dbl[(long)m * 64 + 32 + threadIdx.x];
#endif
    __syncthreads();
    float dv = dtraw[(long)m * 1024 + d] + dtbv;
    dv = (dv > 20.f) ? dv : log1pf(__expf(dv));
    const float xcv = xcf[(long)m * 1024 + d];
    float acc = 0.f;
#pragma unroll
    for (int n = 0; n < 16; ++n) {
      h[n] = __expf(dv * a[n]) * h[n] + dv * sBC[n] * xcv;
      acc += h[n] * sBC[16 + n];
    }
    float y = acc + dsk * xcv;
    const float zv = xz[(long)m * 2048 + 1024 + d];
    y *= zv / (1.f + __expf(-zv));
    yh[(long)m * 1024 + d] = (_Float16)y;
    __syncthreads();
  }
}

// ------------------- residual add + LayerNorm (D=512) -----------------------
__global__ void k_ln(const float* __restrict__ m2o, const float* __restrict__ feats,
                     const float* __restrict__ g, const float* __restrict__ b,
                     float* __restrict__ out) {
  __shared__ float s1[256], s2[256];
  const int m = blockIdx.x, tid = threadIdx.x;
  const long o = (long)m * 512;
  const float v0 = m2o[o + tid] + feats[o + tid];
  const float v1 = m2o[o + 256 + tid] + feats[o + 256 + tid];
  s1[tid] = v0 + v1;
  s2[tid] = v0 * v0 + v1 * v1;
  __syncthreads();
  for (int s = 128; s > 0; s >>= 1) {
    if (tid < s) { s1[tid] += s1[tid + s]; s2[tid] += s2[tid + s]; }
    __syncthreads();
  }
  const float mean = s1[0] * (1.f / 512.f);
  const float var = s2[0] * (1.f / 512.f) - mean * mean;
  const float inv = rsqrtf(var + 1e-5f);
  out[o + tid] = (v0 - mean) * inv * g[tid] + b[tid];
  out[o + 256 + tid] = (v1 - mean) * inv * g[256 + tid] + b[256 + tid];
}

// ---------------------------------------------------------------------------
extern "C" void kernel_launch(void* const* d_in, const int* in_sizes, int n_in,
                              void* d_out, int out_size, void* d_ws, size_t ws_size,
                              hipStream_t stream) {
  const float* x = (const float*)d_in[0];
  const float* c1w = (const float*)d_in[1];
  const float* c1b = (const float*)d_in[2];
  const float* c2w = (const float*)d_in[3];
  const float* c2b = (const float*)d_in[4];
  const float* c3w = (const float*)d_in[5];
  const float* c3b = (const float*)d_in[6];
  const float* fcw = (const float*)d_in[7];
  const float* fcb = (const float*)d_in[8];
  const float* ipw = (const float*)d_in[9];
  const float* c1dw = (const float*)d_in[10];
  const float* c1db = (const float*)d_in[11];
  const float* xpw = (const float*)d_in[12];
  const float* dtw = (const float*)d_in[13];
  const float* dtb = (const float*)d_in[14];
  const float* alog = (const float*)d_in[15];
  const float* dskip = (const float*)d_in[16];
  const float* opw = (const float*)d_in[17];
  const float* m1w = (const float*)d_in[18];
  const float* m1b = (const float*)d_in[19];
  const float* m2w = (const float*)d_in[20];
  const float* m2b = (const float*)d_in[21];
  const float* lng = (const float*)d_in[22];
  const float* lnb = (const float*)d_in[23];
  float* out = (float*)d_out;
  (void)in_sizes; (void)n_in; (void)out_size; (void)ws_size;

  char* base = (char*)d_ws;
  size_t off = 0;
  auto alloc = [&](size_t bytes) -> void* {
    off = (off + 255) & ~(size_t)255;
    void* p = base + off;
    off += bytes;
    return p;
  };
  _Float16* w_c1h = (_Float16*)alloc((size_t)32 * 256 * 2);
  _Float16* w_c2h = (_Float16*)alloc((size_t)64 * 512 * 2);
  _Float16* w_c3h = (_Float16*)alloc((size_t)64 * 576 * 2);
  _Float16* w_fch = (_Float16*)alloc((size_t)512 * 3136 * 2);
  _Float16* w_iph = (_Float16*)alloc((size_t)2048 * 512 * 2);
  _Float16* w_xph = (_Float16*)alloc((size_t)64 * 1024 * 2);
  _Float16* w_dth = (_Float16*)alloc((size_t)1024 * 32 * 2);
  _Float16* w_oph = (_Float16*)alloc((size_t)512 * 1024 * 2);
  _Float16* w_m1h = (_Float16*)alloc((size_t)256 * 512 * 2);
  _Float16* w_m2h = (_Float16*)alloc((size_t)512 * 256 * 2);
  _Float16* a1 = (_Float16*)alloc((size_t)819200 * 32 * 2);
  _Float16* a2 = (_Float16*)alloc((size_t)165888 * 64 * 2);
  _Float16* a3 = (_Float16*)alloc((size_t)100352 * 64 * 2);
  float* feats_f = (float*)alloc((size_t)2048 * 512 * 4);
  _Float16* feats_h = (_Float16*)alloc((size_t)2048 * 512 * 2);
  float* xz_f = (float*)alloc((size_t)2048 * 2048 * 4);
  float* xc_f = (float*)alloc((size_t)2048 * 1024 * 4);
  _Float16* xc_h = (_Float16*)alloc((size_t)2048 * 1024 * 2);
  float* dbl_f = (float*)alloc((size_t)2048 * 64 * 4);
  _Float16* dbl_h = (_Float16*)alloc((size_t)2048 * 64 * 2);
  float* dtraw = (float*)alloc((size_t)2048 * 1024 * 4);
  _Float16* y_h = (_Float16*)alloc((size_t)2048 * 1024 * 2);
  _Float16* mout_h = (_Float16*)alloc((size_t)2048 * 512 * 2);
  _Float16* hid_h = (_Float16*)alloc((size_t)2048 * 256 * 2);
  float* m2o_f = (float*)alloc((size_t)2048 * 512 * 4);

  auto nb = [](long n) { return (unsigned)((n + 255) / 256); };
  auto gb = [](long M, long N, long NT) {  // 8 waves (register tiles) per block
    long tiles = (M / 16) * (N / (16 * NT));
    return (unsigned)((tiles + 7) / 8);
  };

  // --- weight prep ---
  k_cvt<<<nb(32 * 256), 256, 0, stream>>>(c1w, w_c1h, 32 * 256);
  k_perm_c2<<<nb(64 * 512), 256, 0, stream>>>(c2w, w_c2h);
  k_perm_c3<<<nb(64 * 576), 256, 0, stream>>>(c3w, w_c3h);
  k_perm_fc<<<nb(512 * 3136), 256, 0, stream>>>(fcw, w_fch);
  k_cvt<<<nb(2048 * 512), 256, 0, stream>>>(ipw, w_iph, 2048 * 512);
  k_cvt<<<nb(64 * 1024), 256, 0, stream>>>(xpw, w_xph, 64 * 1024);
  k_cvt<<<nb(1024 * 32), 256, 0, stream>>>(dtw, w_dth, 1024 * 32);
  k_cvt<<<nb(512 * 1024), 256, 0, stream>>>(opw, w_oph, 512 * 1024);
  k_cvt<<<nb(256 * 512), 256, 0, stream>>>(m1w, w_m1h, 256 * 512);
  k_cvt<<<nb(512 * 256), 256, 0, stream>>>(m2w, w_m2h, 512 * 256);

  // --- conv stack (implicit GEMM, WMMA, A-fragment reuse across N tiles) ---
  k_conv<true, 2, 4, 8, 84, 84, 20, 20, 4>
      <<<gb(819200, 32, 2), 256, 0, stream>>>(x, w_c1h, c1b, a1, 819200, 32, 256);
  k_conv<false, 4, 32, 4, 20, 20, 9, 9, 2>
      <<<gb(165888, 64, 4), 256, 0, stream>>>(a1, w_c2h, c2b, a2, 165888, 64, 512);
  k_conv<false, 4, 64, 3, 9, 9, 7, 7, 1>
      <<<gb(100352, 64, 4), 256, 0, stream>>>(a2, w_c3h, c3b, a3, 100352, 64, 576);

  // --- fc (relu) -> feats ---
  k_gemm<4><<<gb(2048, 512, 4), 256, 0, stream>>>(a3, 3136, w_fch, 3136, fcb,
                                                  feats_f, feats_h, 2048, 512,
                                                  3136, 1);
  // --- in_proj -> xz (xin | z) ---
  k_gemm<4><<<gb(2048, 2048, 4), 256, 0, stream>>>(feats_h, 512, w_iph, 512,
                                                   nullptr, xz_f, nullptr, 2048,
                                                   2048, 512, 0);
  // --- depthwise causal conv1d + SiLU -> xc ---
  k_conv1d<<<nb((long)2048 * 1024), 256, 0, stream>>>(xz_f, c1dw, c1db, xc_f, xc_h);
  // --- x_proj -> dbl = [dt_raw32 | B16 | C16] ---
  k_gemm<4><<<gb(2048, 64, 4), 256, 0, stream>>>(xc_h, 1024, w_xph, 1024, nullptr,
                                                 dbl_f, dbl_h, 2048, 64, 1024, 0);
  // --- dt_proj (K=32 slice of dbl) ---
  k_gemm<4><<<gb(2048, 1024, 4), 256, 0, stream>>>(dbl_h, 64, w_dth, 32, nullptr,
                                                   dtraw, nullptr, 2048, 1024, 32,
                                                   0);
  // --- selective scan + D-skip + z-gate -> y (fp16) ---
  k_ssm<<<dim3(16, 4), 256, 0, stream>>>(dbl_f, dtraw, dtb, alog, dskip, xc_f,
                                         xz_f, y_h);
  // --- out_proj ---
  k_gemm<4><<<gb(2048, 512, 4), 256, 0, stream>>>(y_h, 1024, w_oph, 1024, nullptr,
                                                  nullptr, mout_h, 2048, 512, 1024,
                                                  0);
  // --- mlp1 (relu), mlp2 ---
  k_gemm<4><<<gb(2048, 256, 4), 256, 0, stream>>>(mout_h, 512, w_m1h, 512, m1b,
                                                  nullptr, hid_h, 2048, 256, 512,
                                                  1);
  k_gemm<4><<<gb(2048, 512, 4), 256, 0, stream>>>(hid_h, 256, w_m2h, 256, m2b,
                                                  m2o_f, nullptr, 2048, 512, 256,
                                                  0);
  // --- residual + LayerNorm -> out (rows m=t*B+b are already (T,B,D) order) ---
  k_ln<<<2048, 256, 0, stream>>>(m2o_f, feats_f, lng, lnb, out);
}
